// HMM_70334384439665
// MI455X (gfx1250) — matrix operand order repaired and checked by reading, not compiled, not run on hardware
//
#include <hip/hip_runtime.h>

typedef __attribute__((ext_vector_type(16))) _Float16 v16h;
typedef __attribute__((ext_vector_type(8)))  float    v8f;
typedef _Float16 half_t;

#define HMM_B 64
#define HMM_T 512
#define HMM_H 512
#define LDA   520   // halves; multiple of 8 -> 16B-aligned rows for ds b128 loads

union FragH { v16h h; unsigned u[8]; uint4 q[2]; };

// ---------------------------------------------------------------------------
// Row-wise logsumexp: one block per row, online (m,s) accumulation.
// ---------------------------------------------------------------------------
__global__ void row_lse_kernel(const float* __restrict__ x,
                               float* __restrict__ out, int cols) {
  const int row = blockIdx.x;
  const float* r = x + (size_t)row * cols;
  float m = -3.0e38f, s = 0.0f;
  for (int j = threadIdx.x; j < cols; j += blockDim.x) {
    float v = r[j];
    if (v > m) { s = s * __expf(m - v) + 1.0f; m = v; }
    else       { s += __expf(v - m); }
  }
  // warp reduce (m,s)
  for (int o = 16; o > 0; o >>= 1) {
    float mo = __shfl_xor(m, o, 32);
    float so = __shfl_xor(s, o, 32);
    float mn = fmaxf(m, mo);
    s = s * __expf(m - mn) + so * __expf(mo - mn);
    m = mn;
  }
  __shared__ float sm[8], ss[8];
  const int w = threadIdx.x >> 5, lane = threadIdx.x & 31;
  if (lane == 0) { sm[w] = m; ss[w] = s; }
  __syncthreads();
  if (threadIdx.x == 0) {
    float M = sm[0], S = ss[0];
    const int nw = blockDim.x >> 5;
    for (int i = 1; i < nw; ++i) {
      float mo = sm[i], so = ss[i];
      float mn = fmaxf(M, mo);
      S = S * __expf(M - mn) + so * __expf(mo - mn);
      M = mn;
    }
    out[row] = M + __logf(S);
  }
}

// ---------------------------------------------------------------------------
// Pack Texp = softmax(tm) (row-wise) into WMMA B-fragment order, f16.
// Tile (kt,nt) covers K=kt*32..+31, N=nt*16..+15. Storage:
//   tp[tile*256 + lane*8 + v] = {Texp[k][n], Texp[k+1][n]} packed halves,
//   k = kt*32 + 16*(lane/16) + 2v, n = nt*16 + lane%16.
// ---------------------------------------------------------------------------
__global__ void pack_tm_kernel(const float* __restrict__ tm,
                               const float* __restrict__ lse,
                               unsigned* __restrict__ tp) {
  const int tid = blockIdx.x * blockDim.x + threadIdx.x;  // H*H/2 total
  const int tile = tid >> 8;
  const int dw   = tid & 255;
  const int lane = dw >> 3;
  const int v    = dw & 7;
  const int kt = tile >> 5;
  const int nt = tile & 31;
  const int k = kt * 32 + ((lane >> 4) << 4) + v * 2;
  const int n = nt * 16 + (lane & 15);
  const float a = __expf(tm[(size_t)k * HMM_H + n] - lse[k]);
  const float b = __expf(tm[(size_t)(k + 1) * HMM_H + n] - lse[k + 1]);
  union { half_t h[2]; unsigned u; } pk;
  pk.h[0] = (half_t)a;
  pk.h[1] = (half_t)b;
  tp[tid] = pk.u;
}

// ---------------------------------------------------------------------------
// Pre-gather log-emissions: emg[b,t,j] = em[j, ids[b,t]] - lse_em[j]  (f16)
// ---------------------------------------------------------------------------
__global__ void gather_em_kernel(const int* __restrict__ ids,
                                 const float* __restrict__ em,
                                 const float* __restrict__ lse_em,
                                 half_t* __restrict__ emg, int V) {
  const size_t idx = (size_t)blockIdx.x * blockDim.x + threadIdx.x;  // B*T*H
  const int j = (int)(idx & (HMM_H - 1));
  const size_t bt = idx >> 9;  // H == 512
  const int id = ids[bt];
  emg[idx] = (half_t)(em[(size_t)j * V + id] - lse_em[j]);
}

// ---------------------------------------------------------------------------
// Main forward scan. One block = 16 batch rows, 512 threads = 16 waves.
// Wave w owns N-tiles {2w, 2w+1} of the per-step [16,512]x[512,512] GEMM.
// alpha kept as f16 exp(alpha - M[b]) in LDS, running f32 offset M[b].
// ---------------------------------------------------------------------------
__global__ __launch_bounds__(512) void hmm_fwd_kernel(
    const float* __restrict__ p, const float* __restrict__ lsep,
    const unsigned* __restrict__ Tpack, const half_t* __restrict__ emg,
    float* __restrict__ out) {
  __shared__ half_t Ash[16 * LDA];
  __shared__ float  wpart[16][16];  // [wave][row] partial max
  __shared__ float  rowmax[16];
  __shared__ float  Mrun[16];
  __shared__ float  resbuf[16];

  const int tid  = threadIdx.x;
  const int w    = tid >> 5;
  const int lane = tid & 31;
  const int b0   = blockIdx.x * 16;

  // ---- t = 0 init: wave w handles batch row b0+w
  {
    const int b = w;
    const float lp = lsep[0];
    const half_t* e0 = emg + (size_t)(b0 + b) * HMM_T * HMM_H;
    float vals[16];
    float mx = -3.0e38f;
#pragma unroll
    for (int q = 0; q < 16; ++q) {
      const int j = lane + 32 * q;
      const float v = p[j] - lp + (float)e0[j];
      vals[q] = v;
      mx = fmaxf(mx, v);
    }
#pragma unroll
    for (int o = 16; o > 0; o >>= 1) mx = fmaxf(mx, __shfl_xor(mx, o, 32));
#pragma unroll
    for (int q = 0; q < 16; ++q) {
      const int j = lane + 32 * q;
      Ash[b * LDA + j] = (half_t)__expf(vals[q] - mx);
    }
    if (lane == 0) Mrun[b] = mx;
  }
  __syncthreads();

  const int nt0  = w * 2;
  const int m    = lane & 15;        // A-fragment row
  const int koff = (lane >> 4) * 16; // A-fragment K offset for upper half-wave
  const int m0   = (lane >> 4) * 8;  // C-fragment row group
  const int jc   = lane & 15;        // C-fragment column within tile
  const int j0   = nt0 * 16 + jc;
  const int j1   = j0 + 16;

  for (int t = 1; t < HMM_T; ++t) {
    v8f c0 = {}; v8f c1 = {};
    for (int kt = 0; kt < 16; ++kt) {
      const int kb = kt * 32;
      FragH af, bf0, bf1;
      // A fragment: 2x ds_load_b128 (16B-aligned; LDA multiple of 8)
      af.q[0] = *(const uint4*)&Ash[m * LDA + kb + koff];
      af.q[1] = *(const uint4*)&Ash[m * LDA + kb + 8 + koff];
      // B fragments: fully coalesced global b128 pairs (resident in L2)
      const uint4* t0 = (const uint4*)(Tpack + (((size_t)(kt * 32 + nt0) * 32 + lane) << 3));
      const uint4* t1 = (const uint4*)(Tpack + (((size_t)(kt * 32 + nt0 + 1) * 32 + lane) << 3));
      bf0.q[0] = t0[0]; bf0.q[1] = t0[1];
      bf1.q[0] = t1[0]; bf1.q[1] = t1[1];
      c0 = __builtin_amdgcn_wmma_f32_16x16x32_f16(false, af.h, false, bf0.h,
                                                  (short)0, c0, false, false);
      c1 = __builtin_amdgcn_wmma_f32_16x16x32_f16(false, af.h, false, bf1.h,
                                                  (short)0, c1, false, false);
    }
    // Epilogue: alpha_new = M[b] + log(C) + emg[b,t,j]; renormalize.
    float v0a[8], v1a[8], pm[8];
    const half_t* ebase = emg + ((size_t)(b0 + m0) * HMM_T + t) * HMM_H;
#pragma unroll
    for (int r = 0; r < 8; ++r) {
      const float e0 = (float)ebase[(size_t)r * HMM_T * HMM_H + j0];
      const float e1 = (float)ebase[(size_t)r * HMM_T * HMM_H + j1];
      const float a0 = __logf(c0[r]) + e0;
      const float a1 = __logf(c1[r]) + e1;
      v0a[r] = a0; v1a[r] = a1;
      pm[r] = fmaxf(a0, a1);
    }
    // per-row max across the 16 column-lanes of each half-wave
#pragma unroll
    for (int o = 1; o < 16; o <<= 1) {
#pragma unroll
      for (int r = 0; r < 8; ++r) pm[r] = fmaxf(pm[r], __shfl_xor(pm[r], o, 32));
    }
    if (jc == 0) {
#pragma unroll
      for (int r = 0; r < 8; ++r) wpart[w][m0 + r] = pm[r];
    }
    __syncthreads();
    if (tid < 16) {
      float rm = wpart[0][tid];
#pragma unroll
      for (int ww = 1; ww < 16; ++ww) rm = fmaxf(rm, wpart[ww][tid]);
      rowmax[tid] = rm;
      Mrun[tid] += rm;
    }
    __syncthreads();
#pragma unroll
    for (int r = 0; r < 8; ++r) {
      const int b = m0 + r;
      const float rm = rowmax[b];
      Ash[b * LDA + j0] = (half_t)__expf(v0a[r] - rm);
      Ash[b * LDA + j1] = (half_t)__expf(v1a[r] - rm);
    }
    __syncthreads();
  }

  // Final: res[b] = M[b] + log(sum_j exp(alphahat[b,j]))
  {
    const int b = w;
    float s = 0.0f;
#pragma unroll
    for (int q = 0; q < 16; ++q) s += (float)Ash[b * LDA + lane + 32 * q];
#pragma unroll
    for (int o = 16; o > 0; o >>= 1) s += __shfl_xor(s, o, 32);
    if (lane == 0) resbuf[b] = Mrun[b] + __logf(s);
  }
  __syncthreads();
  if (tid == 0) {
    float acc = 0.0f;
#pragma unroll
    for (int i = 0; i < 16; ++i) acc += resbuf[i];
    atomicAdd(out, -acc / ((float)HMM_B * (float)HMM_T));
  }
}

// ---------------------------------------------------------------------------
extern "C" void kernel_launch(void* const* d_in, const int* in_sizes, int n_in,
                              void* d_out, int out_size, void* d_ws, size_t ws_size,
                              hipStream_t stream) {
  const int*   ids = (const int*)d_in[0];    // [B,T] int32
  const float* em  = (const float*)d_in[1];  // [H,V]
  const float* tm  = (const float*)d_in[2];  // [H,H]
  const float* p   = (const float*)d_in[3];  // [H]
  const int H = HMM_H;
  const int V = in_sizes[1] / H;  // 32000

  char* ws = (char*)d_ws;
  float*    lse_em = (float*)(ws + 0);      // 512 f32
  float*    lse_tm = (float*)(ws + 2048);   // 512 f32
  float*    lse_p  = (float*)(ws + 4096);   // 1 f32
  unsigned* Tpack  = (unsigned*)(ws + 8192);                  // H*H/2 dwords (512 KB)
  half_t*   emg    = (half_t*)(ws + 8192 + (size_t)H * H * 2); // B*T*H f16 (32 MB)

  hipMemsetAsync(d_out, 0, sizeof(float), stream);

  // Row logsumexps of em, tm, p
  row_lse_kernel<<<H, 256, 0, stream>>>(em, lse_em, V);
  row_lse_kernel<<<H, 256, 0, stream>>>(tm, lse_tm, H);
  row_lse_kernel<<<1, 256, 0, stream>>>(p, lse_p, H);

  // Pack softmax(tm) into WMMA-B fragment layout
  pack_tm_kernel<<<(H * H / 2) / 256, 256, 0, stream>>>(tm, lse_tm, Tpack);

  // Pre-gather emissions for all (b,t): hoists random HBM access out of scan
  {
    const size_t total = (size_t)HMM_B * HMM_T * HMM_H;
    gather_em_kernel<<<(unsigned)(total / 256), 256, 0, stream>>>(ids, em, lse_em, emg, V);
  }

  // Serial forward scan: 4 independent workgroups, 16 batch rows each
  hmm_fwd_kernel<<<HMM_B / 16, 512, 0, stream>>>(p, lse_p, Tpack, emg, (float*)d_out);
}
